// Net_40733469835754
// MI455X (gfx1250) — compile-verified, matrix operand order checked by reading, not run on hardware
//
#include <hip/hip_runtime.h>
#include <math.h>
#include <stdint.h>

typedef __attribute__((ext_vector_type(16))) _Float16 v16h;
typedef __attribute__((ext_vector_type(4)))  _Float16 v4h;
typedef __attribute__((ext_vector_type(8)))  float    v8f;

constexpr int kN  = 20000;
constexpr int kE  = 320000;
constexpr int kNE = 2000;
constexpr int kH  = 4;
constexpr int kD  = 128;
constexpr int kK  = 10;
constexpr int kF  = kH * kD;   // 512

__device__ __forceinline__ int imin(int a, int b) { return a < b ? a : b; }

// ---------------- helpers ----------------
__device__ __forceinline__ int f2oi(float f) {
  int i = __float_as_int(f);
  return (i >= 0) ? i : (i ^ 0x7fffffff);
}
__device__ __forceinline__ float oi2f(int i) {
  return __int_as_float((i >= 0) ? i : (i ^ 0x7fffffff));
}

// Async global->LDS copy of 16 bytes (gfx1250 ASYNCcnt path).
__device__ __forceinline__ void async_ld_b128(unsigned lds_off, const void* gaddr) {
  asm volatile("global_load_async_to_lds_b128 %0, %1, off"
               :: "v"(lds_off), "v"((unsigned long long)(uintptr_t)gaddr)
               : "memory");
}
__device__ __forceinline__ void wait_async0() {
  asm volatile("s_wait_asynccnt 0x0" ::: "memory");
}

// Gather a 16-half WMMA fragment from an LDS tile with row stride 32 halves.
// Layout per CDNA5 ISA 7.12.2 (16-bit A 16x32): lanes 0-15 rows M, K in
// {0..7} (vgpr 0-3) and {16..23} (vgpr 4-7); lanes 16-31: K {8..15},{24..31}.
__device__ __forceinline__ v16h frag_ld(const _Float16* base, int row, int lane) {
  v16h f;
  const int hi = (lane & 16) ? 8 : 0;
  const _Float16* r = base + row * 32;
#pragma unroll
  for (int v = 0; v < 8; ++v) {
    int kb = (v < 4) ? (hi + 2 * v) : (16 + hi + 2 * (v - 4));
    f[2 * v + 0] = r[kb + 0];
    f[2 * v + 1] = r[kb + 1];
  }
  return f;
}

// ---------------- WMMA GEMM:  out[M,Nout] = X[M,K] @ W[Nout,K]^T (+bias) ----
// f16 inputs (pre-converted), f32 accumulate.
// block = 256 thr = 8 waves in 2(M) x 4(N); block tile 64(M) x 256(N);
// wave tile 32x64 -> 8 WMMAs per K-step of 32. Staging via async-to-LDS.
__global__ __launch_bounds__(256) void gemm_f16_wmma(
    const _Float16* __restrict__ X, const _Float16* __restrict__ W,
    const float* __restrict__ bias, float* __restrict__ out,
    int M, int K, int Nout) {
  __shared__ _Float16 As[64 * 32];    // 4 KB
  __shared__ _Float16 Bs[256 * 32];   // 16 KB

  const int t    = threadIdx.x;
  const int lane = t & 31;
  const int wave = t >> 5;
  const int wm   = wave >> 2;  // 0..1
  const int wn   = wave & 3;   // 0..3
  const int mBlk = blockIdx.x * 64;
  const int nBlk = blockIdx.y * 256;

  // LDS byte offsets (aperture occupies addr[63:32]; low word is LDS offset)
  const unsigned asBase = (unsigned)(uintptr_t)(&As[0]);
  const unsigned bsBase = (unsigned)(uintptr_t)(&Bs[0]);

  // staging coords: thread t -> row t/4, col (t%4)*8 halves (16B each)
  const int sRow = t >> 2;
  const int sCol = (t & 3) * 8;
  const int gmA  = imin(mBlk + sRow, M - 1);     // clamp: garbage rows land in
  int gnB[4];                                    // unstored outputs only
#pragma unroll
  for (int r = 0; r < 4; ++r) gnB[r] = imin(nBlk + sRow + 64 * r, Nout - 1);

  v8f z = {};
  v8f acc[2][4];
#pragma unroll
  for (int i = 0; i < 2; ++i)
#pragma unroll
    for (int j = 0; j < 4; ++j) acc[i][j] = z;

  for (int k0 = 0; k0 < K; k0 += 32) {
    async_ld_b128(asBase + (unsigned)(sRow * 32 + sCol) * 2,
                  X + (size_t)gmA * K + k0 + sCol);
#pragma unroll
    for (int r = 0; r < 4; ++r)
      async_ld_b128(bsBase + (unsigned)((sRow + 64 * r) * 32 + sCol) * 2,
                    W + (size_t)gnB[r] * K + k0 + sCol);
    if (k0 + 32 < K)
      __builtin_prefetch(X + (size_t)gmA * K + k0 + 32 + sCol, 0, 1);
    wait_async0();
    __syncthreads();

    v16h a0 = frag_ld(As, wm * 32 + 0  + (lane & 15), lane);
    v16h a1 = frag_ld(As, wm * 32 + 16 + (lane & 15), lane);
#pragma unroll
    for (int nt = 0; nt < 4; ++nt) {
      v16h b = frag_ld(Bs, wn * 64 + nt * 16 + (lane & 15), lane);
      acc[0][nt] = __builtin_amdgcn_wmma_f32_16x16x32_f16(false, a0, false, b,
                                                          (short)0, acc[0][nt], false, false);
      acc[1][nt] = __builtin_amdgcn_wmma_f32_16x16x32_f16(false, a1, false, b,
                                                          (short)0, acc[1][nt], false, false);
    }
    __syncthreads();
  }

  // store: C/D layout -> m = mTile + r + 8*(lane>=16), n = nTile + lane%16
#pragma unroll
  for (int mi = 0; mi < 2; ++mi) {
    const int mBase = mBlk + wm * 32 + mi * 16 + ((lane >> 4) << 3);
#pragma unroll
    for (int nt = 0; nt < 4; ++nt) {
      const int n = nBlk + wn * 64 + nt * 16 + (lane & 15);
      if (n < Nout) {
        const float bv = bias ? bias[n] : 0.f;
#pragma unroll
        for (int r = 0; r < 8; ++r) {
          const int m = mBase + r;
          if (m < M) out[(size_t)m * Nout + n] = acc[mi][nt][r] + bv;
        }
      }
    }
  }
}

// ---------------- f32 -> f16 conversion (vector) ----------------
__global__ void cvt_f16_kernel(const float* __restrict__ src, _Float16* __restrict__ dst,
                               long n) {
  long i4 = ((long)blockIdx.x * blockDim.x + threadIdx.x) * 4;
  if (i4 >= n) return;
  float4 v = *reinterpret_cast<const float4*>(src + i4);
  v4h h = {(_Float16)v.x, (_Float16)v.y, (_Float16)v.z, (_Float16)v.w};
  *reinterpret_cast<v4h*>(dst + i4) = h;
}

// ---------------- small elementwise / graph kernels ----------------
__global__ void concat_kernel(const float* __restrict__ pos, const float* __restrict__ emb,
                              float* __restrict__ h0) {
  int idx = blockIdx.x * blockDim.x + threadIdx.x;
  if (idx >= kN * 128) return;
  int node = idx >> 7, d = idx & 127;
  h0[idx] = (d < 2) ? pos[node * 2 + d] : emb[node * 126 + (d - 2)];
}

__global__ void node_scores_kernel(const float* __restrict__ feat, const float* __restrict__ al,
                                   const float* __restrict__ ar, float* __restrict__ el,
                                   float* __restrict__ er) {
  int idx = blockIdx.x * blockDim.x + threadIdx.x;
  if (idx >= kN * kH) return;
  int node = idx / kH, h = idx % kH;
  const float* f = feat + (size_t)node * kF + h * kD;
  float sl = 0.f, sr = 0.f;
#pragma unroll 4
  for (int d = 0; d < kD; ++d) {
    float v = f[d];
    sl += v * al[h * kD + d];
    sr += v * ar[h * kD + d];
  }
  el[idx] = sl;
  er[idx] = sr;
}

__global__ void fill_int_kernel(int* __restrict__ p, int v, int n) {
  int idx = blockIdx.x * blockDim.x + threadIdx.x;
  if (idx < n) p[idx] = v;
}

__global__ void edge_logits_kernel(const int* __restrict__ src, const int* __restrict__ dst,
                                   const float* __restrict__ el, const float* __restrict__ er,
                                   float* __restrict__ e, int* __restrict__ emaxi) {
  int idx = blockIdx.x * blockDim.x + threadIdx.x;
  if (idx >= kE * kH) return;
  int ed = idx / kH, h = idx % kH;
  int s = src[ed], d = dst[ed];
  float x = el[s * kH + h] + er[d * kH + h];
  x = (x >= 0.f) ? x : 0.2f * x;
  e[idx] = x;
  atomicMax(&emaxi[d * kH + h], f2oi(x));
}

__global__ void emax_fix_kernel(int* __restrict__ emaxi) {
  int idx = blockIdx.x * blockDim.x + threadIdx.x;
  if (idx >= kN * kH) return;
  int v = emaxi[idx];
  float m = (v == (int)0x80000000) ? 0.f : oi2f(v);
  reinterpret_cast<float*>(emaxi)[idx] = m;  // reuse buffer as float
}

__global__ void edge_exp_kernel(const int* __restrict__ dst, const float* __restrict__ emaxf,
                                float* __restrict__ e, float* __restrict__ asum) {
  int idx = blockIdx.x * blockDim.x + threadIdx.x;
  if (idx >= kE * kH) return;
  int ed = idx / kH, h = idx % kH;
  int d = dst[ed];
  float a = expf(e[idx] - emaxf[d * kH + h]);
  e[idx] = a;
  atomicAdd(&asum[d * kH + h], a);
}

__global__ void asum_fix_kernel(float* __restrict__ asum) {
  int idx = blockIdx.x * blockDim.x + threadIdx.x;
  if (idx >= kN * kH) return;
  float v = asum[idx];
  asum[idx] = (v > 0.f) ? v : 1.f;
}

__global__ void edge_scatter_kernel(const int* __restrict__ src, const int* __restrict__ dst,
                                    const float* __restrict__ a, const float* __restrict__ asum,
                                    const float* __restrict__ feat, float* __restrict__ rst) {
  size_t idx = (size_t)blockIdx.x * blockDim.x + threadIdx.x;
  if (idx >= (size_t)kE * kH * 32) return;
  int d4 = idx & 31;
  size_t tmp = idx >> 5;
  int h = tmp % kH;
  int ed = tmp / kH;
  int s = src[ed], dd = dst[ed];
  float alpha = a[ed * kH + h] / asum[dd * kH + h];
  const float* fs = feat + (size_t)s * kF + h * kD + d4 * 4;
  float* rd = rst + (size_t)dd * kF + h * kD + d4 * 4;
#pragma unroll
  for (int c = 0; c < 4; ++c) atomicAdd(&rd[c], alpha * fs[c]);
}

__global__ void combine_relu_kernel(const float* __restrict__ rst, const float* __restrict__ res,
                                    float* __restrict__ hout) {
  size_t idx = (size_t)blockIdx.x * blockDim.x + threadIdx.x;
  if (idx >= (size_t)kN * kF) return;
  float v = rst[idx] + res[idx];
  hout[idx] = fmaxf(v, 0.f);
}

__global__ void mean_heads_kernel(const float* __restrict__ rst, const float* __restrict__ res,
                                  float* __restrict__ hf) {
  int idx = blockIdx.x * blockDim.x + threadIdx.x;
  if (idx >= kN * kD) return;
  int node = idx >> 7, d = idx & 127;
  float s = 0.f;
#pragma unroll
  for (int h = 0; h < kH; ++h) {
    size_t o = (size_t)node * kF + h * kD + d;
    s += rst[o] + res[o];
  }
  hf[idx] = s * 0.25f;
}

__global__ void entity_agg_kernel(const float* __restrict__ hf, const float* __restrict__ pos,
                                  const int* __restrict__ entity, float* __restrict__ esum,
                                  float* __restrict__ psum, float* __restrict__ cnt) {
  int idx = blockIdx.x * blockDim.x + threadIdx.x;
  if (idx >= kN * kD) return;
  int node = idx >> 7, d = idx & 127;
  int e = entity[node];
  atomicAdd(&esum[(size_t)e * kD + d], hf[idx]);
  if (d < 2) atomicAdd(&psum[e * 2 + d], pos[node * 2 + d]);
  if (d == 0) atomicAdd(&cnt[e], 1.f);
}

__global__ void ent_pos_kernel(const float* __restrict__ psum, const float* __restrict__ cnt,
                               float* __restrict__ opos) {
  int idx = blockIdx.x * blockDim.x + threadIdx.x;
  if (idx >= kNE * 2) return;
  opos[idx] = psum[idx] / fmaxf(cnt[idx >> 1], 1.f);
}

__global__ void knn_kernel(const float* __restrict__ epos, int* __restrict__ knn) {
  int i = blockIdx.x * blockDim.x + threadIdx.x;
  if (i >= kNE) return;
  float px = epos[i * 2 + 0], py = epos[i * 2 + 1];
  float bd[kK];
  int bi[kK];
#pragma unroll
  for (int k = 0; k < kK; ++k) { bd[k] = 3.4e38f; bi[k] = 0; }
  for (int j = 0; j < kNE; ++j) {
    float dx = epos[j * 2 + 0] - px;
    float dy = epos[j * 2 + 1] - py;
    float d2 = dx * dx + dy * dy;
    if (d2 < bd[kK - 1]) {
      int k = kK - 1;
      while (k > 0 && d2 < bd[k - 1]) {
        bd[k] = bd[k - 1];
        bi[k] = bi[k - 1];
        --k;
      }
      bd[k] = d2;
      bi[k] = j;
    }
  }
#pragma unroll
  for (int k = 0; k < kK; ++k) knn[i * kK + k] = bi[k];
}

// one wave32 per pair: sigmoid(relu(|a-b| @ W1^T + b1) @ W2^T + b2)
__global__ __launch_bounds__(256) void pair_mlp_kernel(
    const float* __restrict__ feats, const int* __restrict__ sidx,
    const int* __restrict__ didx, int kdiv, const float* __restrict__ W1,
    const float* __restrict__ b1, const float* __restrict__ W2,
    const float* __restrict__ b2, float* __restrict__ out, int npairs) {
  __shared__ float xs[8][kD];
  const int lane = threadIdx.x & 31;
  const int wave = threadIdx.x >> 5;
  const int pair = blockIdx.x * 8 + wave;
  const bool active = pair < npairs;
  int s = 0, d = 0;
  if (active) {
    s = sidx[pair];
    d = (kdiv > 0) ? (pair / kdiv) : didx[pair];
  }
  if (active) {
    const float* fa = feats + (size_t)s * kD;
    const float* fb = feats + (size_t)d * kD;
#pragma unroll
    for (int c = 0; c < 4; ++c) {
      int i = lane + 32 * c;
      xs[wave][i] = fabsf(fa[i] - fb[i]);
    }
  }
  __syncthreads();
  float total = 0.f;
  if (active) {
#pragma unroll
    for (int c = 0; c < 4; ++c) {
      int j = lane + 32 * c;
      float hv = b1[j];
      const float* w = W1 + (size_t)j * kD;
#pragma unroll 4
      for (int i = 0; i < kD; ++i) hv += xs[wave][i] * w[i];
      hv = fmaxf(hv, 0.f);
      total += hv * W2[j];
    }
  }
#pragma unroll
  for (int off = 16; off > 0; off >>= 1) total += __shfl_xor(total, off, 32);
  if (active && lane == 0) out[pair] = 1.f / (1.f + expf(-(total + b2[0])));
}

__global__ void ent_class_kernel(const float* __restrict__ states, const float* __restrict__ Wc,
                                 const float* __restrict__ bc, float* __restrict__ out) {
  int idx = blockIdx.x * blockDim.x + threadIdx.x;
  if (idx >= kNE * 4) return;
  int e = idx >> 2, c = idx & 3;
  const float* st = states + (size_t)e * kD;
  const float* w = Wc + (size_t)c * kD;
  float s = bc[c];
#pragma unroll 4
  for (int i = 0; i < kD; ++i) s += st[i] * w[i];
  out[idx] = 1.f / (1.f + expf(-s));
}

// ---------------- launch ----------------
extern "C" void kernel_launch(void* const* d_in, const int* in_sizes, int n_in,
                              void* d_out, int out_size, void* d_ws, size_t ws_size,
                              hipStream_t stream) {
  const float* position = (const float*)d_in[0];
  const float* w_embed  = (const float*)d_in[1];
  const int*   src      = (const int*)d_in[2];
  const int*   dst      = (const int*)d_in[3];
  const int*   entity   = (const int*)d_in[4];
  const float* Wgat[3]  = {(const float*)d_in[5], (const float*)d_in[9], (const float*)d_in[13]};
  const float* algat[3] = {(const float*)d_in[6], (const float*)d_in[10], (const float*)d_in[14]};
  const float* argat[3] = {(const float*)d_in[7], (const float*)d_in[11], (const float*)d_in[15]};
  const float* Wrgat[3] = {(const float*)d_in[8], (const float*)d_in[12], (const float*)d_in[16]};
  const float* Wg1 = (const float*)d_in[17];
  const float* bg1 = (const float*)d_in[18];
  const float* Wg2 = (const float*)d_in[19];
  const float* bg2 = (const float*)d_in[20];
  const float* We  = (const float*)d_in[21];
  const float* be  = (const float*)d_in[22];
  const float* Wl1 = (const float*)d_in[23];
  const float* bl1 = (const float*)d_in[24];
  const float* Wl2 = (const float*)d_in[25];
  const float* bl2 = (const float*)d_in[26];
  const float* Wc  = (const float*)d_in[27];
  const float* bc  = (const float*)d_in[28];

  // workspace layout (floats, then halves)
  float* h0    = (float*)d_ws;             // N*512 (layer input, layer1 uses N*128)
  float* feat  = h0   + (size_t)kN * kF;   // N*512
  float* res   = feat + (size_t)kN * kF;   // N*512
  float* rst   = res  + (size_t)kN * kF;   // N*512
  float* el    = rst  + (size_t)kN * kF;   // N*H
  float* er    = el   + (size_t)kN * kH;   // N*H
  float* ebuf  = er   + (size_t)kN * kH;   // E*H
  int*   emaxi = (int*)(ebuf + (size_t)kE * kH);   // N*H (reused as float after fix)
  float* asum  = (float*)(emaxi + (size_t)kN * kH);// N*H
  float* hf    = asum + (size_t)kN * kH;   // N*128
  float* esum  = hf   + (size_t)kN * kD;   // NE*128
  float* psum  = esum + (size_t)kNE * kD;  // NE*2
  float* cnt   = psum + (size_t)kNE * 2;   // NE
  float* estat = cnt  + kNE;               // NE*128
  int*   knn   = (int*)(estat + (size_t)kNE * kD); // NE*K
  _Float16* hx16 = (_Float16*)(knn + (size_t)kNE * kK); // N*512
  _Float16* w16a = hx16 + (size_t)kN * kF;              // 512*512
  _Float16* w16b = w16a + (size_t)kF * kF;              // 512*512
  _Float16* es16 = w16b + (size_t)kF * kF;              // NE*128
  _Float16* we16 = es16 + (size_t)kNE * kD;             // 128*128

  // output layout
  float* out_groups = (float*)d_out;            // E
  float* out_class  = out_groups + kE;          // NE*4
  float* out_pos    = out_class + kNE * 4;      // NE*2
  float* out_link   = out_pos + kNE * 2;        // NE*K

  const int T = 256;
  auto blk = [](long n, int t) { return (int)((n + t - 1) / t); };

  // h0 = concat(position, w_embed)  [N,128], then to f16
  concat_kernel<<<blk((long)kN * 128, T), T, 0, stream>>>(position, w_embed, h0);
  cvt_f16_kernel<<<blk((long)kN * kD / 4, T), T, 0, stream>>>(h0, hx16, (long)kN * kD);

  for (int layer = 0; layer < 3; ++layer) {
    const int Fin = (layer == 0) ? 128 : kF;
    cvt_f16_kernel<<<blk((long)kF * Fin / 4, T), T, 0, stream>>>(Wgat[layer], w16a, (long)kF * Fin);
    cvt_f16_kernel<<<blk((long)kF * Fin / 4, T), T, 0, stream>>>(Wrgat[layer], w16b, (long)kF * Fin);

    dim3 ggrid(blk(kN, 64), blk(kF, 256));
    gemm_f16_wmma<<<ggrid, T, 0, stream>>>(hx16, w16a, nullptr, feat, kN, Fin, kF);
    gemm_f16_wmma<<<ggrid, T, 0, stream>>>(hx16, w16b, nullptr, res, kN, Fin, kF);

    node_scores_kernel<<<blk((long)kN * kH, T), T, 0, stream>>>(feat, algat[layer], argat[layer], el, er);

    fill_int_kernel<<<blk((long)kN * kH, T), T, 0, stream>>>(emaxi, (int)0x80000000, kN * kH);
    hipMemsetAsync(asum, 0, (size_t)kN * kH * sizeof(float), stream);
    hipMemsetAsync(rst, 0, (size_t)kN * kF * sizeof(float), stream);

    edge_logits_kernel<<<blk((long)kE * kH, T), T, 0, stream>>>(src, dst, el, er, ebuf, emaxi);
    emax_fix_kernel<<<blk((long)kN * kH, T), T, 0, stream>>>(emaxi);
    edge_exp_kernel<<<blk((long)kE * kH, T), T, 0, stream>>>(dst, (const float*)emaxi, ebuf, asum);
    asum_fix_kernel<<<blk((long)kN * kH, T), T, 0, stream>>>(asum);
    edge_scatter_kernel<<<blk((long)kE * kH * 32, T), T, 0, stream>>>(src, dst, ebuf, asum, feat, rst);

    if (layer < 2) {
      combine_relu_kernel<<<blk((long)kN * kF, T), T, 0, stream>>>(rst, res, h0);
      cvt_f16_kernel<<<blk((long)kN * kF / 4, T), T, 0, stream>>>(h0, hx16, (long)kN * kF);
    } else {
      mean_heads_kernel<<<blk((long)kN * kD, T), T, 0, stream>>>(rst, res, hf);
    }
  }

  // groups_score over edges
  pair_mlp_kernel<<<blk(kE, 8), T, 0, stream>>>(hf, src, dst, 0, Wg1, bg1, Wg2, bg2,
                                                out_groups, kE);

  // entity segment sums
  hipMemsetAsync(esum, 0, (size_t)kNE * kD * sizeof(float), stream);
  hipMemsetAsync(psum, 0, (size_t)kNE * 2 * sizeof(float), stream);
  hipMemsetAsync(cnt, 0, (size_t)kNE * sizeof(float), stream);
  entity_agg_kernel<<<blk((long)kN * kD, T), T, 0, stream>>>(hf, position, entity, esum, psum, cnt);

  // entity_states = esum @ We^T + be   (WMMA GEMM, M=NE, K=128, Nout=128)
  cvt_f16_kernel<<<blk((long)kNE * kD / 4, T), T, 0, stream>>>(esum, es16, (long)kNE * kD);
  cvt_f16_kernel<<<blk((long)kD * kD / 4, T), T, 0, stream>>>(We, we16, (long)kD * kD);
  {
    dim3 ggrid(blk(kNE, 64), blk(kD, 256));
    gemm_f16_wmma<<<ggrid, T, 0, stream>>>(es16, we16, be, estat, kNE, kD, kD);
  }

  // entity centroids -> output + knn graph
  ent_pos_kernel<<<blk((long)kNE * 2, T), T, 0, stream>>>(psum, cnt, out_pos);
  knn_kernel<<<blk(kNE, 128), 128, 0, stream>>>(out_pos, knn);

  // entity link scores
  pair_mlp_kernel<<<blk((long)kNE * kK, 8), T, 0, stream>>>(estat, knn, nullptr, kK,
                                                            Wl1, bl1, Wl2, bl2,
                                                            out_link, kNE * kK);

  // entity class
  ent_class_kernel<<<blk((long)kNE * 4, T), T, 0, stream>>>(estat, Wc, bc, out_class);
}
